// Net_35708358099625
// MI455X (gfx1250) — matrix operand order repaired and checked by reading, not compiled, not run on hardware
//
#include <hip/hip_runtime.h>
#include <math.h>

#define NN 100000
#define NE 1600000
#define NG 64
#define EMB 64
#define HID 128
#define NSTEPS 6
#define NET 3

typedef __attribute__((ext_vector_type(16))) _Float16 v16h;
typedef __attribute__((ext_vector_type(8)))  _Float16 h8;
typedef __attribute__((ext_vector_type(4)))  _Float16 h4;
typedef __attribute__((ext_vector_type(8)))  float    v8f;

// ---------------- WMMA helpers ----------------
// A operand (16x32 f16): lane l -> row M = l&15, hi = l>>4.
// element e = 2j+half -> K = ((j>>2)<<4) + (hi<<3) + ((j&3)<<1) + half
// => elements 0..7  are K = koff +      hi*8 + 0..7   (8 consecutive halves)
//    elements 8..15 are K = koff + 16 + hi*8 + 0..7   (8 consecutive halves)

// A tile from an f32 row (converts on the fly)
__device__ __forceinline__ v16h load_a_f32(const float* __restrict__ rowp, int koff) {
  int hi = (threadIdx.x >> 4) & 1;
  v16h a;
#pragma unroll
  for (int j = 0; j < 8; ++j) {
    int k = koff + ((j >> 2) << 4) + (hi << 3) + ((j & 3) << 1);
    float2 v = *(const float2*)(rowp + k);
    a[2 * j]     = (_Float16)v.x;
    a[2 * j + 1] = (_Float16)v.y;
  }
  return a;
}

// A tile straight from an f16 row: two b128 loads, no conversion
__device__ __forceinline__ v16h load_a_f16(const _Float16* __restrict__ rowp, int koff) {
  int hi = (threadIdx.x >> 4) & 1;
  h8 lo = *(const h8*)(rowp + koff + hi * 8);
  h8 up = *(const h8*)(rowp + koff + 16 + hi * 8);
  v16h a;
#pragma unroll
  for (int i = 0; i < 8; ++i) { a[i] = lo[i]; a[8 + i] = up[i]; }
  return a;
}

// B operand (32x16 f16): lane l -> col N = l&15; elements 0..7 K = hi*8+0..7, elements 8..15 K = 16+hi*8+0..7 (same as A).
// B[k][n] = W[n][k] (row-major [out][in] weight) -> two runs of 8 f16 of W row n, at koff+hi*8 and koff+16+hi*8.
__device__ __forceinline__ v16h load_b_f16(const _Float16* __restrict__ W, int ldk, int nrow, int koff) {
  int hi = (threadIdx.x >> 4) & 1;
  const _Float16* p = W + (size_t)nrow * ldk + koff + hi * 8;
  h8 lo = *(const h8*)p;
  h8 up = *(const h8*)(p + 16);
  v16h b;
#pragma unroll
  for (int i = 0; i < 8; ++i) { b[i] = lo[i]; b[8 + i] = up[i]; }
  return b;
}

__device__ __forceinline__ v8f wmma16(v16h a, v16h b, v8f c) {
  return __builtin_amdgcn_wmma_f32_16x16x32_f16(false, a, false, b, (short)0, c, false, false);
}

__device__ __forceinline__ v8f vzero8() {
  v8f c;
#pragma unroll
  for (int i = 0; i < 8; ++i) c[i] = 0.f;
  return c;
}

__device__ __forceinline__ float fsigmoid(float x) { return 1.f / (1.f + __expf(-x)); }
__device__ __forceinline__ float ftanh(float x)    { return 1.f - 2.f / (1.f + __expf(2.f * x)); }

// ---------------- kernels ----------------

__global__ void k_f32_to_f16(const float* __restrict__ s, _Float16* __restrict__ d, int n) {
  int i = blockIdx.x * 256 + threadIdx.x;
  if (i < n) d[i] = (_Float16)s[i];
}

// h0 = embed[type_ids] @ W_red^T + b_red ; h = h0 (f32 + f16 shadow). grid(3125), block 64
__global__ void k_embed(const int* __restrict__ type_ids, const float* __restrict__ emb,
                        const _Float16* __restrict__ Wred16, const float* __restrict__ bred,
                        float* __restrict__ h0, float* __restrict__ h,
                        _Float16* __restrict__ h16) {
  int wave = threadIdx.x >> 5, lane = threadIdx.x & 31;
  int nodeBase = (blockIdx.x * 2 + wave) * 16;
  int mrow = lane & 15, hi = (lane >> 4) & 1;
  const float* arow = emb + (size_t)type_ids[nodeBase + mrow] * EMB;
  v8f acc[8];
#pragma unroll
  for (int t = 0; t < 8; ++t) acc[t] = vzero8();
#pragma unroll
  for (int kb = 0; kb < 2; ++kb) {
    v16h A = load_a_f32(arow, kb * 32);
#pragma unroll
    for (int t = 0; t < 8; ++t) {
      v16h B = load_b_f16(Wred16, EMB, t * 16 + mrow, kb * 32);
      acc[t] = wmma16(A, B, acc[t]);
    }
  }
#pragma unroll
  for (int t = 0; t < 8; ++t) {
    int col = t * 16 + mrow;
    float bias = bred[col];
#pragma unroll
    for (int r = 0; r < 8; ++r) {
      int row = nodeBase + r + 8 * hi;
      float v = acc[t][r] + bias;
      h0[(size_t)row * HID + col]  = v;
      h[(size_t)row * HID + col]   = v;
      h16[(size_t)row * HID + col] = (_Float16)v;
    }
  }
}

// trans[k] = h @ W_msg[k]^T + b_msg[k] (f16 in, f16 out). grid(3125, 3), block 64
__global__ void k_msg(const _Float16* __restrict__ h16, const _Float16* __restrict__ Wmsg16,
                      const float* __restrict__ bmsg, _Float16* __restrict__ trans) {
  int wave = threadIdx.x >> 5, lane = threadIdx.x & 31;
  int nodeBase = (blockIdx.x * 2 + wave) * 16;
  int k = blockIdx.y;
  int mrow = lane & 15, hi = (lane >> 4) & 1;
  const _Float16* arow = h16 + (size_t)(nodeBase + mrow) * HID;
  const _Float16* W = Wmsg16 + (size_t)k * HID * HID;
  v8f acc[8];
#pragma unroll
  for (int t = 0; t < 8; ++t) acc[t] = vzero8();
#pragma unroll
  for (int kb = 0; kb < 4; ++kb) {
    v16h A = load_a_f16(arow, kb * 32);
#pragma unroll
    for (int t = 0; t < 8; ++t) {
      v16h B = load_b_f16(W, HID, t * 16 + mrow, kb * 32);
      acc[t] = wmma16(A, B, acc[t]);
    }
  }
  _Float16* outk = trans + (size_t)k * NN * HID;
#pragma unroll
  for (int t = 0; t < 8; ++t) {
    int col = t * 16 + mrow;
    float bias = bmsg[k * HID + col];
#pragma unroll
    for (int r = 0; r < 8; ++r) {
      int row = nodeBase + r + 8 * hi;
      outk[(size_t)row * HID + col] = (_Float16)(acc[t][r] + bias);
    }
  }
}

// a[dst] += trans[etype, src] (f16 gather, f32 atomic scatter). one wave per edge.
__global__ void k_edge(const _Float16* __restrict__ trans, const int* __restrict__ src,
                       const int* __restrict__ dst, const int* __restrict__ et,
                       float* __restrict__ acc) {
  long long gid = (long long)blockIdx.x * blockDim.x + threadIdx.x;
  int e = (int)(gid >> 5);
  int lane = (int)(gid & 31);
  if (e >= NE) return;
  int s = src[e], d = dst[e], k = et[e];
  h4 v = *(const h4*)(trans + ((size_t)k * NN + s) * HID + lane * 4);
  float* out = acc + (size_t)d * HID + lane * 4;
  atomicAdd(out + 0, (float)v[0]);
  atomicAdd(out + 1, (float)v[1]);
  atomicAdd(out + 2, (float)v[2]);
  atomicAdd(out + 3, (float)v[3]);
}

// fused GRU: hout = GRU(a, h). grid(3125, 8 col-chunks), block 64. 6 WMMA tiles per K-step.
__global__ void k_gru(const float* __restrict__ a, const float* __restrict__ h,
                      const _Float16* __restrict__ h16,
                      const _Float16* __restrict__ Wih16, const float* __restrict__ bih,
                      const _Float16* __restrict__ Whh16, const float* __restrict__ bhh,
                      float* __restrict__ hout, _Float16* __restrict__ hout16) {
  int wave = threadIdx.x >> 5, lane = threadIdx.x & 31;
  int nodeBase = (blockIdx.x * 2 + wave) * 16;
  int c = blockIdx.y;
  int mrow = lane & 15, hi = (lane >> 4) & 1;
  const float*    ax = a   + (size_t)(nodeBase + mrow) * HID;
  const _Float16* ah = h16 + (size_t)(nodeBase + mrow) * HID;
  v8f ir = vzero8(), hr = vzero8(), iz = vzero8(), hz = vzero8(), in_ = vzero8(), hn = vzero8();
  int wr = c * 16 + mrow;  // weight row == output column for this lane
#pragma unroll
  for (int kb = 0; kb < 4; ++kb) {
    int ko = kb * 32;
    v16h Ax = load_a_f32(ax, ko);
    v16h Ah = load_a_f16(ah, ko);
    v16h B;
    B = load_b_f16(Wih16, HID, wr, ko);        ir  = wmma16(Ax, B, ir);
    B = load_b_f16(Whh16, HID, wr, ko);        hr  = wmma16(Ah, B, hr);
    B = load_b_f16(Wih16, HID, 128 + wr, ko);  iz  = wmma16(Ax, B, iz);
    B = load_b_f16(Whh16, HID, 128 + wr, ko);  hz  = wmma16(Ah, B, hz);
    B = load_b_f16(Wih16, HID, 256 + wr, ko);  in_ = wmma16(Ax, B, in_);
    B = load_b_f16(Whh16, HID, 256 + wr, ko);  hn  = wmma16(Ah, B, hn);
  }
  int col = c * 16 + mrow;
  float b_ir = bih[col],        b_hr = bhh[col];
  float b_iz = bih[128 + col],  b_hz = bhh[128 + col];
  float b_in = bih[256 + col],  b_hn = bhh[256 + col];
#pragma unroll
  for (int r = 0; r < 8; ++r) {
    int row = nodeBase + r + 8 * hi;
    float rr = fsigmoid(ir[r] + b_ir + hr[r] + b_hr);
    float zz = fsigmoid(iz[r] + b_iz + hz[r] + b_hz);
    float nn = ftanh(in_[r] + b_in + rr * (hn[r] + b_hn));
    float ho = h[(size_t)row * HID + col];
    float hv = (1.f - zz) * nn + zz * ho;
    hout[(size_t)row * HID + col]   = hv;
    hout16[(size_t)row * HID + col] = (_Float16)hv;
  }
}

__global__ void k_zero(float4* __restrict__ p, int n4) {
  int i = blockIdx.x * 256 + threadIdx.x;
  if (i < n4) p[i] = make_float4(0.f, 0.f, 0.f, 0.f);
}

// gate[n] = dot(concat(h,h0), wg) + bg. one wave per node (8 waves/block).
__global__ void k_gate(const float* __restrict__ h, const float* __restrict__ h0,
                       const float* __restrict__ wg, const float* __restrict__ bg,
                       float* __restrict__ gate) {
  int wave = threadIdx.x >> 5, lane = threadIdx.x & 31;
  int n = blockIdx.x * 8 + wave;
  float4 hv = *(const float4*)(h + (size_t)n * HID + lane * 4);
  float4 gv = *(const float4*)(h0 + (size_t)n * HID + lane * 4);
  float4 w1 = *(const float4*)(wg + lane * 4);
  float4 w2 = *(const float4*)(wg + 128 + lane * 4);
  float acc = hv.x * w1.x + hv.y * w1.y + hv.z * w1.z + hv.w * w1.w +
              gv.x * w2.x + gv.y * w2.y + gv.z * w2.z + gv.w * w2.w;
#pragma unroll
  for (int off = 16; off > 0; off >>= 1) acc += __shfl_xor(acc, off, 32);
  if (lane == 0) gate[n] = acc + bg[0];
}

__global__ void k_init_pool(float* __restrict__ m, float* __restrict__ ssum,
                            float* __restrict__ ro) {
  int i = blockIdx.x * 256 + threadIdx.x;
  if (i < NG) m[i] = -__builtin_inff();
  else if (i < 2 * NG) ssum[i - NG] = 0.f;
  if (i < NG * 256) ro[i] = 0.f;
}

__global__ void k_segmax(const float* __restrict__ gate, const int* __restrict__ gid,
                         float* __restrict__ m) {
  int i = blockIdx.x * 256 + threadIdx.x;
  if (i >= NN) return;
  float v = gate[i];
  float* addr = m + gid[i];
  if (v >= 0.f) atomicMax((int*)addr, __float_as_int(v));
  else          atomicMin((unsigned int*)addr, __float_as_uint(v));
}

__global__ void k_expsum(const float* __restrict__ gate, const int* __restrict__ gid,
                         const float* __restrict__ m, float* __restrict__ ssum,
                         float* __restrict__ ebuf) {
  int i = blockIdx.x * 256 + threadIdx.x;
  if (i >= NN) return;
  int g = gid[i];
  float e = __expf(gate[i] - m[g]);
  ebuf[i] = e;
  atomicAdd(&ssum[g], e);
}

// readout[g] += alpha_n * feat_n. one thread per (node, channel).
__global__ void k_readout(const float* __restrict__ h, const float* __restrict__ h0,
                          const float* __restrict__ ebuf, const float* __restrict__ ssum,
                          const int* __restrict__ gid, float* __restrict__ ro) {
  int t = blockIdx.x * 256 + threadIdx.x;
  int n = t >> 8, c = t & 255;
  int g = gid[n];
  float alpha = ebuf[n] / ssum[g];
  float f = (c < HID) ? h[(size_t)n * HID + c] : h0[(size_t)n * HID + (c - HID)];
  atomicAdd(&ro[g * 256 + c], alpha * f);
}

// out[g] = dot(readout[g], W_out) + b_out. one wave per graph.
__global__ void k_out(const float* __restrict__ ro, const float* __restrict__ wout,
                      const float* __restrict__ bout, float* __restrict__ out) {
  int wave = threadIdx.x >> 5, lane = threadIdx.x & 31;
  int g = blockIdx.x * 2 + wave;
  const float* p = ro + g * 256 + lane * 8;
  const float* w = wout + lane * 8;
  float acc = 0.f;
#pragma unroll
  for (int i = 0; i < 8; ++i) acc += p[i] * w[i];
#pragma unroll
  for (int off = 16; off > 0; off >>= 1) acc += __shfl_xor(acc, off, 32);
  if (lane == 0) out[g] = acc + bout[0];
}

// ---------------- driver ----------------
extern "C" void kernel_launch(void* const* d_in, const int* in_sizes, int n_in,
                              void* d_out, int out_size, void* d_ws, size_t ws_size,
                              hipStream_t stream) {
  (void)in_sizes; (void)n_in; (void)out_size; (void)ws_size;
  const int*   type_ids = (const int*)d_in[0];
  const int*   src      = (const int*)d_in[1];
  const int*   dst      = (const int*)d_in[2];
  const int*   etypes   = (const int*)d_in[3];
  const int*   gids     = (const int*)d_in[4];
  const float* emb      = (const float*)d_in[5];
  const float* Wred     = (const float*)d_in[6];
  const float* bred     = (const float*)d_in[7];
  const float* Wmsg     = (const float*)d_in[8];
  const float* bmsg     = (const float*)d_in[9];
  const float* Wih      = (const float*)d_in[10];
  const float* bih      = (const float*)d_in[11];
  const float* Whh      = (const float*)d_in[12];
  const float* bhh      = (const float*)d_in[13];
  const float* Wgate    = (const float*)d_in[14];
  const float* bgate    = (const float*)d_in[15];
  const float* Wout     = (const float*)d_in[16];
  const float* bout     = (const float*)d_in[17];
  float* out = (float*)d_out;

  // ---- workspace layout ----
  float*    h0    = (float*)d_ws;                              // N*HID f32
  float*    hA    = h0 + (size_t)NN * HID;                     // N*HID f32
  float*    hB    = hA + (size_t)NN * HID;                     // N*HID f32
  float*    acc   = hB + (size_t)NN * HID;                     // N*HID f32
  _Float16* trans = (_Float16*)(acc + (size_t)NN * HID);       // 3*N*HID f16
  _Float16* hA16  = trans + (size_t)NET * NN * HID;            // N*HID f16
  _Float16* hB16  = hA16 + (size_t)NN * HID;                   // N*HID f16
  _Float16* Wred16 = hB16 + (size_t)NN * HID;                  // 128*64
  _Float16* Wmsg16 = Wred16 + HID * EMB;                       // 3*128*128
  _Float16* Wih16  = Wmsg16 + NET * HID * HID;                 // 384*128
  _Float16* Whh16  = Wih16 + 3 * HID * HID;                    // 384*128
  float*    gate  = (float*)(Whh16 + 3 * HID * HID);           // N
  float*    ebuf  = gate + NN;                                 // N
  float*    m     = ebuf + NN;                                 // NG
  float*    ssum  = m + NG;                                    // NG
  float*    ro    = ssum + NG;                                 // NG*256

  const int nodeBlocks2 = (NN / 16) / 2;  // 3125 (2 waves of 16 nodes per 64-thread block)

  // one-time per launch: weights -> f16
  k_f32_to_f16<<<dim3((HID * EMB + 255) / 256), dim3(256), 0, stream>>>(Wred, Wred16, HID * EMB);
  k_f32_to_f16<<<dim3((NET * HID * HID + 255) / 256), dim3(256), 0, stream>>>(Wmsg, Wmsg16, NET * HID * HID);
  k_f32_to_f16<<<dim3((3 * HID * HID + 255) / 256), dim3(256), 0, stream>>>(Wih, Wih16, 3 * HID * HID);
  k_f32_to_f16<<<dim3((3 * HID * HID + 255) / 256), dim3(256), 0, stream>>>(Whh, Whh16, 3 * HID * HID);

  k_embed<<<dim3(nodeBlocks2), dim3(64), 0, stream>>>(type_ids, emb, Wred16, bred, h0, hA, hA16);

  float*    hcur   = hA;
  float*    hnxt   = hB;
  _Float16* hcur16 = hA16;
  _Float16* hnxt16 = hB16;
  for (int step = 0; step < NSTEPS; ++step) {
    k_msg<<<dim3(nodeBlocks2, NET), dim3(64), 0, stream>>>(hcur16, Wmsg16, bmsg, trans);
    k_zero<<<dim3((NN * HID / 4) / 256), dim3(256), 0, stream>>>((float4*)acc, NN * HID / 4);
    k_edge<<<dim3((int)(((long long)NE * 32) / 256)), dim3(256), 0, stream>>>(trans, src, dst, etypes, acc);
    k_gru<<<dim3(nodeBlocks2, 8), dim3(64), 0, stream>>>(acc, hcur, hcur16, Wih16, bih, Whh16, bhh, hnxt, hnxt16);
    float* tf = hcur; hcur = hnxt; hnxt = tf;
    _Float16* th = hcur16; hcur16 = hnxt16; hnxt16 = th;
  }
  // after 6 steps hcur == hA

  k_gate<<<dim3(NN / 8), dim3(256), 0, stream>>>(hcur, h0, Wgate, bgate, gate);
  k_init_pool<<<dim3((NG * 256) / 256), dim3(256), 0, stream>>>(m, ssum, ro);
  k_segmax<<<dim3((NN + 255) / 256), dim3(256), 0, stream>>>(gate, gids, m);
  k_expsum<<<dim3((NN + 255) / 256), dim3(256), 0, stream>>>(gate, gids, m, ssum, ebuf);
  k_readout<<<dim3(NN), dim3(256), 0, stream>>>(hcur, h0, ebuf, ssum, gids, ro);
  k_out<<<dim3(NG / 2), dim3(64), 0, stream>>>(ro, Wout, bout, out);
}
